// AreaAttention_66932770341193
// MI455X (gfx1250) — compile-verified, matrix operand order checked by reading, not run on hardware
//
#include <hip/hip_runtime.h>
#include <hip/hip_bf16.h>

// ---------------------------------------------------------------------------
// AreaAttention on MI455X (gfx1250).
// All matmuls via v_wmma_f32_16x16x32_bf16 (fp32 accumulate).
// Layout prep makes every hot inner loop pure b128 loads + WMMA; softmax
// cross-lane reductions use DPP ROW_XMASK (VALU) instead of ds_bpermute.
// Dims: B=4, H=W=64, C=256, heads=8, head_dim=32, AREA=4.
// ---------------------------------------------------------------------------

typedef __attribute__((ext_vector_type(16))) __bf16 v16bf;
typedef __attribute__((ext_vector_type(8)))  float  v8f;

#define DIM       256
#define QKDIM     512
#define NUM_HEADS 8
#define HEAD_DIM  32
#define AREA_N    4
#define SEQ       4096
#define BATCH     4
#define HH        64
#define WW        64
#define NROWS     (BATCH * SEQ)   /* 16384 token rows */
#define EPS_BN    1e-3f
#define ATTN_SCALE 0.17677669529663687f   /* 1/sqrt(32) */

// 8 dwords viewed as a 16-element bf16 WMMA fragment.
union FragU {
    unsigned u[8];
    v16bf    v;
};

static __device__ __forceinline__ unsigned short f2bf_bits(float f) {
    union { __bf16 h; unsigned short s; } c;
    c.h = (__bf16)f;
    return c.s;
}

// DPP16 ROW_XMASK lane exchange within a 16-lane row (gfx10+ ctrl 0x160|mask):
// matches the WMMA C-layout where each matrix row lives in one 16-lane half.
template <int CTRL>
static __device__ __forceinline__ float dppf(float x) {
    return __int_as_float(__builtin_amdgcn_update_dpp(
        0, __float_as_int(x), CTRL, 0xF, 0xF, true));
}
static __device__ __forceinline__ float row_max16(float x) {
    x = fmaxf(x, dppf<0x161>(x));   // xor 1
    x = fmaxf(x, dppf<0x162>(x));   // xor 2
    x = fmaxf(x, dppf<0x164>(x));   // xor 4
    x = fmaxf(x, dppf<0x168>(x));   // xor 8
    return x;
}
static __device__ __forceinline__ float row_sum16(float x) {
    x += dppf<0x161>(x);
    x += dppf<0x162>(x);
    x += dppf<0x164>(x);
    x += dppf<0x168>(x);
    return x;
}

// K-pair base index for the 16-bit A/B fragment (ISA 7.12.2):
// VGPR j (lanes 0-15): K = {2j,2j+1} for j<4, {16+2(j-4),...} for j>=4;
// lanes 16-31 add +8.  Always even -> packed pair = one aligned dword,
// and j=0..3 / j=4..7 are contiguous dwords (compiler fuses into b128).
static __device__ __forceinline__ int kpair(int j, int lane) {
    int kb = (j < 4) ? (2 * j) : (16 + 2 * (j - 4));
    return kb + ((lane >> 4) << 3);
}

// ---------------------------------------------------------------------------
// Prep kernels (one-shot, tiny vs. the 192MB L2).
// ---------------------------------------------------------------------------
__global__ void __launch_bounds__(256)
cvt_bf16(const float* __restrict__ in, unsigned short* __restrict__ out, int n)
{
    int i = blockIdx.x * 256 + threadIdx.x;
    if (i < n) out[i] = f2bf_bits(in[i]);
}

__global__ void __launch_bounds__(256)
transpose_w_bf16(const float* __restrict__ in,     // K x N, f32
                 unsigned short* __restrict__ out, // N x K, bf16
                 int K, int N)
{
    int i = blockIdx.x * 256 + threadIdx.x;
    if (i >= K * N) return;
    int k = i / N, n = i % N;
    out[(size_t)n * K + k] = f2bf_bits(in[i]);
}

// out = bf16(a + b), elementwise (attn + pe before the proj GEMM)
__global__ void __launch_bounds__(256)
add_cvt_bf16(const float* __restrict__ a, const float* __restrict__ b,
             unsigned short* __restrict__ out, int n)
{
    int i = blockIdx.x * 256 + threadIdx.x;
    if (i < n) out[i] = f2bf_bits(a[i] + b[i]);
}

// ---------------------------------------------------------------------------
// GEMM: A (M x K, bf16 row-major) * Wt^T (Wt is N x K bf16) with fused BN.
// Each wave computes 64(M) x 16(N): 4 accumulators, B-fragment reused 4x.
// Per 32-wide K-chunk: 2 B b128-loads + 8 A b128-loads -> 4 WMMAs.
// MODE 0: write bf16 token-major; q columns (col<256) pre-scaled by 1/sqrt(d)
// MODE 1: write f32 + bf16 [bah][dim][key] transpose (v path)
// MODE 2: write f32 (proj path)
// ---------------------------------------------------------------------------
template <int MODE>
__global__ void __launch_bounds__(256)
gemm_bn_wmma(const unsigned short* __restrict__ Abf,
             const unsigned short* __restrict__ Wt,
             const float* __restrict__ gamma,
             const float* __restrict__ beta,
             const float* __restrict__ mean,
             const float* __restrict__ var,
             float* __restrict__ outF,
             unsigned short* __restrict__ outBF,
             unsigned short* __restrict__ vtOut,
             int K, int N)
{
    const int lane = threadIdx.x & 31;
    const int wid  = threadIdx.x >> 5;
    const int row0 = blockIdx.x * 64;            // M supertile (64 rows)
    const int nt   = blockIdx.y * 8 + wid;       // N tile (16 cols)
    if (nt * 16 >= N) return;                    // wave-uniform guard

    const int hl  = lane & 15;
    const int hi8 = (lane >> 4) << 3;
    const int col = nt * 16 + hl;

    v8f acc[4];
    #pragma unroll
    for (int s = 0; s < 4; ++s)
        acc[s] = (v8f){0.f, 0.f, 0.f, 0.f, 0.f, 0.f, 0.f, 0.f};

    const unsigned short* wrow = Wt + (size_t)col * K;

    for (int k0 = 0; k0 < K; k0 += 32) {
        FragU bw;
        #pragma unroll
        for (int j = 0; j < 8; ++j)
            bw.u[j] = *(const unsigned*)(wrow + k0 + kpair(j, lane));

        #pragma unroll
        for (int s = 0; s < 4; ++s) {
            const unsigned short* arow =
                Abf + (size_t)(row0 + s * 16 + hl) * K + k0;
            FragU a;
            #pragma unroll
            for (int j = 0; j < 8; ++j)
                a.u[j] = *(const unsigned*)(arow + kpair(j, lane));
            acc[s] = __builtin_amdgcn_wmma_f32_16x16x32_bf16(
                       false, a.v, false, bw.v, (short)0, acc[s], false, false);
        }
    }

    // BN epilogue: y = (x - m) * (g * rsqrt(v + eps)) + b
    float sc = gamma[col] * rsqrtf(var[col] + EPS_BN);
    float sh = beta[col] - mean[col] * sc;
    if constexpr (MODE == 0) {
        // fold the attention 1/sqrt(head_dim) into the q half (cols < 256)
        const float qs = (col < DIM) ? ATTN_SCALE : 1.f;
        sc *= qs;
        sh *= qs;
    }

    #pragma unroll
    for (int s = 0; s < 4; ++s) {
        const int rb = row0 + s * 16 + hi8;
        #pragma unroll
        for (int i = 0; i < 8; ++i) {
            const int   row = rb + i;
            const float y   = acc[s][i] * sc + sh;
            if constexpr (MODE == 0) {
                outBF[(size_t)row * N + col] = f2bf_bits(y);
            } else if constexpr (MODE == 1) {
                outF[(size_t)row * N + col] = y;
                // vt[((b*4+area)*8+head)*32 + dim][key]
                const int head = col >> 5, dim = col & 31;
                const size_t vti =
                    ((size_t)(((row >> 10) * 8 + head) * 32 + dim)) * 1024
                    + (row & 1023);
                vtOut[vti] = f2bf_bits(y);
            } else {
                outF[(size_t)row * N + col] = y;
            }
        }
    }
}

// ---------------------------------------------------------------------------
// 5x5 depthwise conv on v (NHWC, f32) + BN -> pe.
// ---------------------------------------------------------------------------
__global__ void __launch_bounds__(256)
dwconv5_bn(const float* __restrict__ V,
           const float* __restrict__ Wpe,
           const float* __restrict__ gamma,
           const float* __restrict__ beta,
           const float* __restrict__ mean,
           const float* __restrict__ var,
           float* __restrict__ pe)
{
    const int c = threadIdx.x;
    const int p = blockIdx.x;                 // b*H*W + h*W + w
    const int w = p % WW;
    const int h = (p / WW) % HH;
    const int b = p / (HH * WW);
    const float* vb = V + (size_t)b * HH * WW * DIM;

    float acc = 0.f;
    #pragma unroll
    for (int dh = 0; dh < 5; ++dh) {
        const int hh = h + dh - 2;
        if (hh < 0 || hh >= HH) continue;
        #pragma unroll
        for (int dw = 0; dw < 5; ++dw) {
            const int ww2 = w + dw - 2;
            if (ww2 < 0 || ww2 >= WW) continue;
            acc += vb[((size_t)hh * WW + ww2) * DIM + c] *
                   Wpe[(dh * 5 + dw) * DIM + c];
        }
    }
    const float s = gamma[c] * rsqrtf(var[c] + EPS_BN);
    pe[(size_t)p * DIM + c] = (acc - mean[c]) * s + beta[c];
}

// ---------------------------------------------------------------------------
// Flash-style area attention. grid.x = B*AREA*HEADS (128), grid.y = 16.
// Block = 128 threads = 4 waves; each wave owns a 16-query tile and streams
// 1024 keys in 64-chunks: 4 score WMMAs + 4 P@V WMMAs per iteration.
// q is pre-scaled by 1/sqrt(d) in the qk GEMM. Online-softmax stats live in
// the WMMA C-layout half-wave replication; row reductions are DPP ROW_XMASK;
// P transposes C->A layout via a per-wave LDS tile.
// ---------------------------------------------------------------------------
__global__ void __launch_bounds__(128)
area_attn_wmma(const unsigned short* __restrict__ QKbf,
               const unsigned short* __restrict__ Vt,
               float* __restrict__ O)
{
    __shared__ unsigned short sP[4][16][64];   // per-wave P tile (16q x 64k)

    const int lane = threadIdx.x & 31;
    const int wid  = threadIdx.x >> 5;
    const int bah  = blockIdx.x;               // ((b*4+area)*8+head)
    const int head = bah & 7;
    const int rbase = (bah >> 3) * 1024;       // token-row base of this area
    const int q0    = blockIdx.y * 64 + wid * 16;
    const int hl    = lane & 15;
    const int hi8   = (lane >> 4) << 3;

    // Q fragment (head_dim = 32 = exactly one bf16 WMMA K-chunk)
    FragU aq;
    {
        const unsigned short* qrow =
            QKbf + (size_t)(rbase + q0 + hl) * QKDIM + head * HEAD_DIM;
        #pragma unroll
        for (int j = 0; j < 8; ++j)
            aq.u[j] = *(const unsigned*)(qrow + kpair(j, lane));
    }
    const unsigned short* vtb = Vt + (size_t)bah * HEAD_DIM * 1024;

    float m8[8], l8[8];
    v8f o0 = {0.f, 0.f, 0.f, 0.f, 0.f, 0.f, 0.f, 0.f};
    v8f o1 = {0.f, 0.f, 0.f, 0.f, 0.f, 0.f, 0.f, 0.f};
    #pragma unroll
    for (int i = 0; i < 8; ++i) { m8[i] = -1e30f; l8[i] = 0.f; }

    for (int kc = 0; kc < 1024; kc += 64) {
        // K fragments for key tiles kc + 16*t, t = 0..3
        FragU bk0, bk1, bk2, bk3;
        {
            const unsigned short* kp =
                QKbf + (size_t)(rbase + kc + hl) * QKDIM + DIM + head * HEAD_DIM;
            #pragma unroll
            for (int j = 0; j < 8; ++j) {
                const int kb = kpair(j, lane);
                bk0.u[j] = *(const unsigned*)(kp + kb);
                bk1.u[j] = *(const unsigned*)(kp + (size_t)16 * QKDIM + kb);
                bk2.u[j] = *(const unsigned*)(kp + (size_t)32 * QKDIM + kb);
                bk3.u[j] = *(const unsigned*)(kp + (size_t)48 * QKDIM + kb);
            }
        }
        const v8f z = {0.f, 0.f, 0.f, 0.f, 0.f, 0.f, 0.f, 0.f};
        v8f s0 = __builtin_amdgcn_wmma_f32_16x16x32_bf16(
                   false, aq.v, false, bk0.v, (short)0, z, false, false);
        v8f s1 = __builtin_amdgcn_wmma_f32_16x16x32_bf16(
                   false, aq.v, false, bk1.v, (short)0, z, false, false);
        v8f s2 = __builtin_amdgcn_wmma_f32_16x16x32_bf16(
                   false, aq.v, false, bk2.v, (short)0, z, false, false);
        v8f s3 = __builtin_amdgcn_wmma_f32_16x16x32_bf16(
                   false, aq.v, false, bk3.v, (short)0, z, false, false);

        // online softmax over this 64-key slab (scores already scaled)
        #pragma unroll
        for (int i = 0; i < 8; ++i) {
            const float a0 = s0[i], a1 = s1[i], a2 = s2[i], a3 = s3[i];
            float pm = row_max16(fmaxf(fmaxf(a0, a1), fmaxf(a2, a3)));
            const float nm  = fmaxf(m8[i], pm);
            const float rsc = __expf(m8[i] - nm);
            const float e0 = __expf(a0 - nm);
            const float e1 = __expf(a1 - nm);
            const float e2 = __expf(a2 - nm);
            const float e3 = __expf(a3 - nm);
            const float rs = row_sum16((e0 + e1) + (e2 + e3));
            l8[i] = l8[i] * rsc + rs;
            m8[i] = nm;
            o0[i] *= rsc;
            o1[i] *= rsc;
            const int r = i + hi8;
            sP[wid][r][hl]      = f2bf_bits(e0);
            sP[wid][r][16 + hl] = f2bf_bits(e1);
            sP[wid][r][32 + hl] = f2bf_bits(e2);
            sP[wid][r][48 + hl] = f2bf_bits(e3);
        }
        __syncthreads();

        // P tile back in A layout: keys [0,32) and [32,64)
        FragU ap0, ap1;
        #pragma unroll
        for (int j = 0; j < 8; ++j) {
            const int kb = kpair(j, lane);
            ap0.u[j] = *(const unsigned*)&sP[wid][hl][kb];
            ap1.u[j] = *(const unsigned*)&sP[wid][hl][32 + kb];
        }

        // V fragments from the [dim][key] transpose: pairs are contiguous
        FragU bv00, bv01, bv10, bv11;
        #pragma unroll
        for (int j = 0; j < 8; ++j) {
            const int kb = kpair(j, lane);
            const unsigned short* v0 = vtb + (size_t)hl * 1024 + kc;
            const unsigned short* v1 = vtb + (size_t)(16 + hl) * 1024 + kc;
            bv00.u[j] = *(const unsigned*)(v0 + kb);
            bv01.u[j] = *(const unsigned*)(v0 + 32 + kb);
            bv10.u[j] = *(const unsigned*)(v1 + kb);
            bv11.u[j] = *(const unsigned*)(v1 + 32 + kb);
        }
        o0 = __builtin_amdgcn_wmma_f32_16x16x32_bf16(
               false, ap0.v, false, bv00.v, (short)0, o0, false, false);
        o0 = __builtin_amdgcn_wmma_f32_16x16x32_bf16(
               false, ap1.v, false, bv01.v, (short)0, o0, false, false);
        o1 = __builtin_amdgcn_wmma_f32_16x16x32_bf16(
               false, ap0.v, false, bv10.v, (short)0, o1, false, false);
        o1 = __builtin_amdgcn_wmma_f32_16x16x32_bf16(
               false, ap1.v, false, bv11.v, (short)0, o1, false, false);
        __syncthreads();
    }

    #pragma unroll
    for (int i = 0; i < 8; ++i) {
        const float inv = 1.f / l8[i];
        const int r = rbase + q0 + i + hi8;
        O[(size_t)r * DIM + head * HEAD_DIM + hl]      = o0[i] * inv;
        O[(size_t)r * DIM + head * HEAD_DIM + 16 + hl] = o1[i] * inv;
    }
}

// ---------------------------------------------------------------------------
extern "C" void kernel_launch(void* const* d_in, const int* in_sizes, int n_in,
                              void* d_out, int out_size, void* d_ws, size_t ws_size,
                              hipStream_t stream)
{
    (void)in_sizes; (void)n_in; (void)out_size; (void)ws_size;

    const float* x      = (const float*)d_in[0];
    const float* w_qk   = (const float*)d_in[1];
    const float* g_qk   = (const float*)d_in[2];
    const float* b_qk   = (const float*)d_in[3];
    const float* m_qk   = (const float*)d_in[4];
    const float* v_qk   = (const float*)d_in[5];
    const float* w_v    = (const float*)d_in[6];
    const float* g_v    = (const float*)d_in[7];
    const float* b_v    = (const float*)d_in[8];
    const float* m_v    = (const float*)d_in[9];
    const float* v_v    = (const float*)d_in[10];
    const float* w_pe   = (const float*)d_in[11];
    const float* g_pe   = (const float*)d_in[12];
    const float* b_pe   = (const float*)d_in[13];
    const float* m_pe   = (const float*)d_in[14];
    const float* v_pe   = (const float*)d_in[15];
    const float* w_proj = (const float*)d_in[16];
    const float* g_proj = (const float*)d_in[17];
    const float* b_proj = (const float*)d_in[18];
    const float* m_proj = (const float*)d_in[19];
    const float* v_proj = (const float*)d_in[20];

    // ---- workspace carve-up (all 256B aligned) ----
    char* p = (char*)d_ws;
    auto carve = [&](size_t bytes) {
        char* r = p;
        p += (bytes + 255) & ~(size_t)255;
        return r;
    };
    unsigned short* xbf    = (unsigned short*)carve((size_t)NROWS * DIM * 2);
    unsigned short* wqkT   = (unsigned short*)carve((size_t)QKDIM * DIM * 2);
    unsigned short* wvT    = (unsigned short*)carve((size_t)DIM * DIM * 2);
    unsigned short* wprojT = (unsigned short*)carve((size_t)DIM * DIM * 2);
    unsigned short* qkbf   = (unsigned short*)carve((size_t)NROWS * QKDIM * 2);
    unsigned short* vt     = (unsigned short*)carve((size_t)NROWS * DIM * 2);
    unsigned short* sumbf  = (unsigned short*)carve((size_t)NROWS * DIM * 2);
    float*          vbuf   = (float*)carve((size_t)NROWS * DIM * 4);
    float*          pebuf  = (float*)carve((size_t)NROWS * DIM * 4);
    float*          abuf   = (float*)carve((size_t)NROWS * DIM * 4);

    const dim3 blk256(256);

    // 0) layout/precision prep
    cvt_bf16<<<dim3(NROWS * DIM / 256), blk256, 0, stream>>>(
        x, xbf, NROWS * DIM);
    transpose_w_bf16<<<dim3(DIM * QKDIM / 256), blk256, 0, stream>>>(
        w_qk, wqkT, DIM, QKDIM);
    transpose_w_bf16<<<dim3(DIM * DIM / 256), blk256, 0, stream>>>(
        w_v, wvT, DIM, DIM);
    transpose_w_bf16<<<dim3(DIM * DIM / 256), blk256, 0, stream>>>(
        w_proj, wprojT, DIM, DIM);

    // 1) qk = BN(x @ w_qk) -> bf16 token-major, q half pre-scaled
    gemm_bn_wmma<0><<<dim3(NROWS / 64, QKDIM / 128), blk256, 0, stream>>>(
        xbf, wqkT, g_qk, b_qk, m_qk, v_qk,
        nullptr, qkbf, nullptr, DIM, QKDIM);

    // 2) v = BN(x @ w_v) -> f32 (for conv) + bf16 [bah][dim][key] (for P@V)
    gemm_bn_wmma<1><<<dim3(NROWS / 64, DIM / 128), blk256, 0, stream>>>(
        xbf, wvT, g_v, b_v, m_v, v_v,
        vbuf, nullptr, vt, DIM, DIM);

    // 3) pe = BN(dwconv5x5(v))
    dwconv5_bn<<<dim3(NROWS), dim3(DIM), 0, stream>>>(
        vbuf, w_pe, g_pe, b_pe, m_pe, v_pe, pebuf);

    // 4) area attention -> abuf (f32)
    area_attn_wmma<<<dim3(BATCH * AREA_N * NUM_HEADS, 16), dim3(128), 0, stream>>>(
        qkbf, vt, abuf);

    // 5) sumbf = bf16(attn + pe)
    add_cvt_bf16<<<dim3(NROWS * DIM / 256), blk256, 0, stream>>>(
        abuf, pebuf, sumbf, NROWS * DIM);

    // 6) out = BN(sum @ w_proj) -> f32 d_out
    gemm_bn_wmma<2><<<dim3(NROWS / 64, DIM / 128), blk256, 0, stream>>>(
        sumbf, wprojT, g_proj, b_proj, m_proj, v_proj,
        (float*)d_out, nullptr, nullptr, DIM, DIM);
}
